// SparseMiddleExtractor_72962904425002
// MI455X (gfx1250) — compile-verified, matrix operand order checked by reading, not run on hardware
//
#include <hip/hip_runtime.h>

// ---------------------------------------------------------------------------
// SparseMiddleExtractor on MI455X (gfx1250): implicit-GEMM conv3d chain using
// v_wmma_f32_16x16x32_bf16. Channels-last bf16 activations, f32 accumulate.
// Each wave computes a 32-position (2 rows x 16 x) x 64-channel tile so every
// B fragment load feeds 8 WMMAs.
// ---------------------------------------------------------------------------

typedef __attribute__((ext_vector_type(16))) __bf16 v16bf;
typedef __attribute__((ext_vector_type(8)))  __bf16 v8bf;
typedef __attribute__((ext_vector_type(8)))  float  v8f;

#define HDIM 128
#define WDIM 128
#define HWSZ (HDIM * WDIM)

union AFrag {
    struct { v8bf lo, hi; } s;
    v16bf v;
};

// ---------------------------------------------------------------------------
// Pack conv weights [Co, Ci, KD, KH, KW] f32 -> per-lane WMMA B fragments:
// layout [tap][kb][nt][lane][16] bf16, matching the 16-bit B VGPR layout
// (lane<16: K = {0..7, 16..23}+kb*32 ; lane>=16: K = {8..15, 24..31}+kb*32,
//  N = nt*16 + (lane&15)).
// ---------------------------------------------------------------------------
__global__ void pack_weights_kernel(const float* __restrict__ W,
                                    __bf16* __restrict__ wp,
                                    int Ci, int KD, int KH, int KW) {
    const int  KB    = Ci >> 5;
    const long total = (long)KD * KH * KW * KB * 4L * 32L * 16L;
    long t = (long)blockIdx.x * blockDim.x + threadIdx.x;
    if (t >= total) return;

    int j    = (int)(t & 15);
    int lane = (int)((t >> 4) & 31);
    int nt   = (int)((t >> 9) & 3);
    int kb   = (int)((t >> 11) % KB);
    int tap  = (int)(t / (2048L * KB));

    int td  = tap / (KH * KW);
    int rem = tap % (KH * KW);
    int th  = rem / KW;
    int tw  = rem % KW;

    int n  = nt * 16 + (lane & 15);
    int h8 = (lane >= 16) ? 8 : 0;
    int kl = (j < 8) ? (j + h8) : (j + 8 + h8);
    int ci = kb * 32 + kl;

    float v = W[(((long)n * Ci + ci) * KD + td) * (long)(KH * KW) + th * KW + tw];
    wp[t] = (__bf16)v;
}

// ---------------------------------------------------------------------------
// NCDHW f32 feat * (mask>0.5)  ->  channels-last bf16 [pos][128].
// ---------------------------------------------------------------------------
__global__ void pack_input_kernel(const float* __restrict__ feat,
                                  const float* __restrict__ mask,
                                  __bf16* __restrict__ x0, int D) {
    long t = (long)blockIdx.x * blockDim.x + threadIdx.x;
    long npos = (long)D * HWSZ;
    if (t >= npos * 16) return;
    int  c8  = (int)(t & 15);
    long pos = t >> 4;
    float mv = (mask[pos] > 0.5f) ? 1.0f : 0.0f;
    v8bf o;
#pragma unroll
    for (int i = 0; i < 8; ++i) {
        int c = c8 * 8 + i;
        o[i] = (__bf16)(feat[(long)c * npos + pos] * mv);
    }
    *reinterpret_cast<v8bf*>(x0 + pos * 128 + c8 * 8) = o;
}

// ---------------------------------------------------------------------------
// Downsampled active-set mask: max over dz in {0,1,2} of (mask>0.5).
// ---------------------------------------------------------------------------
__global__ void mask_down_kernel(const float* __restrict__ mask,
                                 float* __restrict__ m1, int Dout) {
    long t = (long)blockIdx.x * blockDim.x + threadIdx.x;
    if (t >= (long)Dout * HWSZ) return;
    int  z1 = (int)(t / HWSZ);
    long p  = t % HWSZ;
    float m = 0.0f;
#pragma unroll
    for (int dz = 0; dz < 3; ++dz)
        if (mask[(long)(2 * z1 + dz) * HWSZ + p] > 0.5f) m = 1.0f;
    m1[t] = m;
}

// ---------------------------------------------------------------------------
// Shifted-GEMM conv via WMMA.
//   grid  = Dout * (HDIM/2) blocks; block = 256 threads = 8 wave32.
//   Wave w owns x in [16w, 16w+16) for output rows y0 and y0+1, all 64 ch.
// Per tap (dz,dy,dx), per k-block: B(32x16 per N-tile) loaded once feeds
// 2 rows x 4 N-tiles = 8 WMMAs.
//   KB       : Cin/32 (4 for Cin=128, 2 for Cin=64)
//   SPATIAL  : true -> 3x3 taps in (h,w) with SAME pad; false -> 1x1
//   MASKMODE : 0 none, 1 raw mask (>0.5), 2 precomputed {0,1} float
//   OUTF32   : false -> channels-last bf16, true -> NCDHW f32
// ---------------------------------------------------------------------------
template <int KB, bool SPATIAL, int MASKMODE, bool OUTF32>
__global__ __launch_bounds__(256)
void conv_wmma_kernel(const __bf16* __restrict__ x,
                      const __bf16* __restrict__ wp,
                      const float* __restrict__ mask,
                      __bf16* __restrict__ outBf, float* __restrict__ outF32,
                      int Din, int Dout, int zStride, int zPad) {
    constexpr int Cin = KB * 32;
    constexpr int khw = SPATIAL ? 3 : 1;
    constexpr int pyx = SPATIAL ? 1 : 0;
    constexpr int ntapsYX = khw * khw;

    const int lane = threadIdx.x & 31;
    const int wave = threadIdx.x >> 5;
    const int z    = blockIdx.x / (HDIM / 2);
    const int y0   = (blockIdx.x % (HDIM / 2)) * 2;
    const int x0w  = wave * 16;

    const int r  = lane & 15;
    const int h8 = (lane >> 4) << 3;   // 0 for lanes 0-15, 8 for lanes 16-31

    v8f acc[2][4] = {{v8f{}, v8f{}, v8f{}, v8f{}},
                     {v8f{}, v8f{}, v8f{}, v8f{}}};

    int tap = 0;
    for (int dz = 0; dz < 3; ++dz) {
        const int zin = z * zStride + dz - zPad;
        if (zin < 0 || zin >= Din) { tap += ntapsYX; continue; }
        const __bf16* __restrict__ xz = x + (long)zin * HDIM * WDIM * Cin;
        for (int dy = 0; dy < khw; ++dy) {
            const int  yin0  = y0 + dy - pyx;        // input row for out row 0
            const bool vy[2] = {(yin0 >= 0) && (yin0 < HDIM),
                                (yin0 + 1 >= 0) && (yin0 + 1 < HDIM)};
            const __bf16* __restrict__ xrow0 = xz + (long)yin0 * WDIM * Cin;
            for (int dx = 0; dx < khw; ++dx, ++tap) {
                const int  xin = x0w + r + dx - pyx;
                const bool inx = (xin >= 0) && (xin < WDIM);
                const __bf16* __restrict__ pa0 = xrow0 + (long)xin * Cin + h8;
                const __bf16* __restrict__ pb =
                    wp + (long)tap * KB * 2048 + lane * 16;

                // Prefetch next tap's packed weights into the WGP cache.
                __builtin_prefetch(pb + (long)KB * 2048, 0, 1);

                // Stage A fragments for both output rows.
                AFrag a[2][KB];
#pragma unroll
                for (int rr = 0; rr < 2; ++rr) {
                    const __bf16* par = pa0 + (long)rr * WDIM * Cin;
                    const bool ok = inx && vy[rr];
#pragma unroll
                    for (int kb = 0; kb < KB; ++kb) {
                        a[rr][kb].v = v16bf{};
                        if (ok) {
                            a[rr][kb].s.lo =
                                *reinterpret_cast<const v8bf*>(par + kb * 32);
                            a[rr][kb].s.hi =
                                *reinterpret_cast<const v8bf*>(par + kb * 32 + 16);
                        }
                    }
                }

                // Per k-block: load B once, feed 2 rows x 4 N-tiles.
#pragma unroll
                for (int kb = 0; kb < KB; ++kb) {
                    v16bf b[4];
#pragma unroll
                    for (int nt = 0; nt < 4; ++nt)
                        b[nt] = *reinterpret_cast<const v16bf*>(
                            pb + (long)kb * 2048 + nt * 512);
#pragma unroll
                    for (int rr = 0; rr < 2; ++rr)
#pragma unroll
                        for (int nt = 0; nt < 4; ++nt)
                            acc[rr][nt] = __builtin_amdgcn_wmma_f32_16x16x32_bf16(
                                false, a[rr][kb].v, false, b[nt], (short)0,
                                acc[rr][nt], false, false);
                }
            }
        }
    }

    // Epilogue: mask, ReLU, store (per output row).
#pragma unroll
    for (int rr = 0; rr < 2; ++rr) {
        const int  yo = y0 + rr;
        const long zy = (long)z * HDIM + yo;
#pragma unroll
        for (int j = 0; j < 8; ++j) {
            const int m  = j + ((lane >= 16) ? 8 : 0);
            const int xo = x0w + m;
            float mv = 1.0f;
            if (MASKMODE == 1)
                mv = (mask[zy * WDIM + xo] > 0.5f) ? 1.0f : 0.0f;
            else if (MASKMODE == 2)
                mv = mask[zy * WDIM + xo];
            if (OUTF32) {
#pragma unroll
                for (int nt = 0; nt < 4; ++nt) {
                    float v = fmaxf(acc[rr][nt][j] * mv, 0.0f);
                    const long co = nt * 16 + (lane & 15);
                    outF32[((co * Dout + z) * (long)HWSZ) +
                           (long)yo * WDIM + xo] = v;
                }
            } else {
                __bf16* po = outBf + (zy * WDIM + xo) * 64;
#pragma unroll
                for (int nt = 0; nt < 4; ++nt) {
                    float v = fmaxf(acc[rr][nt][j] * mv, 0.0f);
                    po[nt * 16 + (lane & 15)] = (__bf16)v;
                }
            }
        }
    }
}

// ---------------------------------------------------------------------------
extern "C" void kernel_launch(void* const* d_in, const int* in_sizes, int n_in,
                              void* d_out, int out_size, void* d_ws,
                              size_t ws_size, hipStream_t stream) {
    (void)in_sizes; (void)n_in; (void)out_size; (void)ws_size;

    const float* feat = (const float*)d_in[0];
    const float* mask = (const float*)d_in[1];
    const float* W0   = (const float*)d_in[2];  // [64,128,3,3,3]
    const float* W1   = (const float*)d_in[3];  // [64, 64,3,1,1]
    const float* Wa   = (const float*)d_in[4];  // [64, 64,3,3,3]
    const float* Wb   = (const float*)d_in[5];  // [64, 64,3,3,3]
    const float* W2   = (const float*)d_in[6];  // [64, 64,3,1,1]
    float* out = (float*)d_out;                 // [64, 9, 128, 128]

    const int D0 = 41, D1 = 20, D2 = 9;

    size_t off = 0;
    auto take = [&](size_t bytes) -> char* {
        char* p = (char*)d_ws + off;
        off = (off + bytes + 255) & ~(size_t)255;
        return p;
    };

    __bf16* x0  = (__bf16*)take((size_t)D0 * HWSZ * 128 * 2);
    __bf16* y0  = (__bf16*)take((size_t)D0 * HWSZ * 64 * 2);
    __bf16* x1  = (__bf16*)take((size_t)D1 * HWSZ * 64 * 2);
    __bf16* x2  = (__bf16*)take((size_t)D1 * HWSZ * 64 * 2);
    float*  m1  = (float*) take((size_t)D1 * HWSZ * 4);
    __bf16* w0p = (__bf16*)take(27UL * 4 * 2048 * 2 + 256);  // +prefetch slack
    __bf16* w1p = (__bf16*)take(3UL  * 2 * 2048 * 2 + 256);
    __bf16* wap = (__bf16*)take(27UL * 2 * 2048 * 2 + 256);
    __bf16* wbp = (__bf16*)take(27UL * 2 * 2048 * 2 + 256);
    __bf16* w2p = (__bf16*)take(3UL  * 2 * 2048 * 2 + 256);

    // ---- weight packing ----
    {
        long n;
        n = 27L * 4 * 2048;
        pack_weights_kernel<<<(n + 255) / 256, 256, 0, stream>>>(W0, w0p, 128, 3, 3, 3);
        n = 3L * 2 * 2048;
        pack_weights_kernel<<<(n + 255) / 256, 256, 0, stream>>>(W1, w1p, 64, 3, 1, 1);
        n = 27L * 2 * 2048;
        pack_weights_kernel<<<(n + 255) / 256, 256, 0, stream>>>(Wa, wap, 64, 3, 3, 3);
        pack_weights_kernel<<<(n + 255) / 256, 256, 0, stream>>>(Wb, wbp, 64, 3, 3, 3);
        n = 3L * 2 * 2048;
        pack_weights_kernel<<<(n + 255) / 256, 256, 0, stream>>>(W2, w2p, 64, 3, 1, 1);
    }

    // ---- input packing (mask applied) + downsampled mask ----
    {
        long n = (long)D0 * HWSZ * 16;
        pack_input_kernel<<<(n + 255) / 256, 256, 0, stream>>>(feat, mask, x0, D0);
        long nm = (long)D1 * HWSZ;
        mask_down_kernel<<<(nm + 255) / 256, 256, 0, stream>>>(mask, m1, D1);
    }

    // ---- conv chain (grid = Dout * HDIM/2: 2 output rows per block) ----
    // subm0: 128->64, 3x3x3 SAME, mask(raw), relu
    conv_wmma_kernel<4, true, 1, false><<<D0 * (HDIM / 2), 256, 0, stream>>>(
        x0, w0p, mask, y0, nullptr, D0, D0, 1, 1);
    // down1: 64->64, (3,1,1) stride(2,1,1) VALID, relu
    conv_wmma_kernel<2, false, 0, false><<<D1 * (HDIM / 2), 256, 0, stream>>>(
        y0, w1p, nullptr, x1, nullptr, D0, D1, 2, 0);
    // subm1a: 64->64, 3x3x3 SAME, mask(m1), relu
    conv_wmma_kernel<2, true, 2, false><<<D1 * (HDIM / 2), 256, 0, stream>>>(
        x1, wap, m1, x2, nullptr, D1, D1, 1, 1);
    // subm1b: 64->64, 3x3x3 SAME, mask(m1), relu  (reuse x1 buffer)
    conv_wmma_kernel<2, true, 2, false><<<D1 * (HDIM / 2), 256, 0, stream>>>(
        x2, wbp, m1, x1, nullptr, D1, D1, 1, 1);
    // down2: 64->64, (3,1,1) stride(2,1,1) VALID, relu -> f32 NCDHW output
    conv_wmma_kernel<2, false, 0, true><<<D2 * (HDIM / 2), 256, 0, stream>>>(
        x1, w2p, nullptr, nullptr, out, D1, D2, 2, 0);
}
